// T5Attention_89936615178581
// MI455X (gfx1250) — matrix operand order repaired and checked by reading, not compiled, not run on hardware
//
#include <hip/hip_runtime.h>
#include <hip/hip_bf16.h>
#include <math.h>

// Problem constants (fixed by the reference).
#define S_LEN   2048
#define BATCH   2
#define NHEADS  16
#define DKV     64
#define DMODEL  1024
#define MROWS   (BATCH * S_LEN)   // 4096 token rows
#define NDELTA  (2 * S_LEN - 1)   // 4095 relative positions

// Padded LDS row strides (u16 units): 16B-aligned, conflict-free dword strides.
#define KROW    80   // kbuf row: 64 u16 payload + pad  (160 B, dword stride 40)
#define VROW    40   // vbuf row: 32 u16 payload + pad  ( 80 B, dword stride 20)
#define PROW    40   // P tile row                      ( 80 B)
#define BTROW   40   // GEMM staged W row               ( 80 B)

typedef unsigned short u16;
typedef __attribute__((ext_vector_type(16))) __bf16 bf16x16;
typedef __attribute__((ext_vector_type(8)))  float  f32x8;

// One WMMA 16x32 (A) or 32x16 (B) bf16 fragment = 16 bf16 per lane = 32 bytes.
union Frag16 {
  bf16x16 bf;
  u16     u[16];
  uint4   q[2];   // two 16-byte chunks -> b128 loads
};

__device__ __forceinline__ u16 f2bf(float f) {
  unsigned u = __float_as_uint(f);
  return (u16)((u + 0x7FFFu + ((u >> 16) & 1u)) >> 16);  // RNE
}

__device__ __forceinline__ f32x8 wmma_bf16(const Frag16& a, const Frag16& b, f32x8 c) {
  return __builtin_amdgcn_wmma_f32_16x16x32_bf16(
      /*neg_a=*/false, a.bf, /*neg_b=*/false, b.bf,
      /*c_mod=*/(short)0, c, /*reuse_a=*/false, /*reuse_b=*/false);
}

// Low 32 bits of a generic pointer into LDS == LDS byte offset (ISA 10.2).
__device__ __forceinline__ unsigned lds_off(const void* p) {
  return (unsigned)(uintptr_t)p;
}

// CDNA5 async copy: global -> LDS, 16 bytes per lane, tracked by ASYNCcnt.
__device__ __forceinline__ void async_b128(unsigned lds, unsigned long long gaddr) {
  asm volatile("global_load_async_to_lds_b128 %0, %1, off"
               :: "v"(lds), "v"(gaddr) : "memory");
}
// Wait for all but the 4 most recently issued async ops (double-buffer wait).
__device__ __forceinline__ void wait_async4() {
  asm volatile("s_wait_asynccnt 0x4" ::: "memory");
}
__device__ __forceinline__ void wait_async0() {
  asm volatile("s_wait_asynccnt 0x0" ::: "memory");
}

// xor-lane reductions via ds_swizzle (immediate pattern, no index VGPR).
// offset = {xor[4:0], or[4:0], and[4:0]} -> SWAPX1/2/4/8.
#define SWZ(x, imm) __int_as_float(__builtin_amdgcn_ds_swizzle(__float_as_int(x), imm))
__device__ __forceinline__ float rowmax16(float v) {
  v = fmaxf(v, SWZ(v, 0x041F));
  v = fmaxf(v, SWZ(v, 0x081F));
  v = fmaxf(v, SWZ(v, 0x101F));
  v = fmaxf(v, SWZ(v, 0x201F));
  return v;   // max over each 16-lane half
}

// ---------------------------------------------------------------------------
// f32 -> bf16 conversion (grid-stride)
// ---------------------------------------------------------------------------
__global__ void cvt_f32_to_bf16(const float* __restrict__ src,
                                u16* __restrict__ dst, int n) {
  int i = blockIdx.x * blockDim.x + threadIdx.x;
  int stride = gridDim.x * blockDim.x;
  for (; i < n; i += stride) dst[i] = f2bf(src[i]);
}

// ---------------------------------------------------------------------------
// Transposed bias table: biasT[h][delta] = rel_emb[bucket(delta)][h]
// ---------------------------------------------------------------------------
__global__ void build_bias(const float* __restrict__ rel_emb,
                           float* __restrict__ biasT) {
  int d = blockIdx.x * blockDim.x + threadIdx.x;   // 0 .. 4094
  int h = blockIdx.y;
  if (d >= NDELTA) return;
  int delta = d - (S_LEN - 1);    // k - q  (== mem - ctx)
  int n = -delta;                 // reference: n = -relative_position
  int ret = (n < 0) ? 16 : 0;     // num_buckets/2
  n = (n < 0) ? -n : n;
  int v;
  if (n < 8) {                    // max_exact = 8
    v = n;
  } else {
    v = 8 + (int)(logf((float)n * 0.125f) * 2.8853900817779268f);
    v = (v > 15) ? 15 : v;
  }
  biasT[(size_t)h * NDELTA + d] = rel_emb[(ret + v) * NHEADS + h];
}

// ---------------------------------------------------------------------------
// GEMM: C[M=4096, N=1024] = A_bf16[M,1024] @ W_bf16[1024,1024]
// Block = 8 waves, block tile 128x64: each wave a 16x64 C strip -> 4 WMMAs per
// staged 32-K slab. Double-buffered LDS stage (one barrier per slab); next
// slab's global W loads overlap this slab's WMMAs. Padded rows kill conflicts.
// OUTMODE 0: bf16 row-major. 1: bf16 scattered to V^T [b,h,d,s]. 2: f32.
// ---------------------------------------------------------------------------
template <int OUTMODE>
__global__ void __launch_bounds__(256)
gemm_bf16(const u16* __restrict__ A, const u16* __restrict__ W,
          u16* __restrict__ outB, float* __restrict__ outF) {
  __shared__ u16 bt[2][64 * BTROW];    // [buf][n][k_local], padded rows
  const int tid  = threadIdx.x;
  const int wave = tid >> 5;
  const int lane = tid & 31;
  const int half = lane >> 4;
  const int l16  = lane & 15;
  const int m0 = blockIdx.x * 128 + wave * 16;
  const int n0 = blockIdx.y * 64;

  f32x8 acc[4] = {f32x8{}, f32x8{}, f32x8{}, f32x8{}};
  const u16* arow = A + (size_t)(m0 + l16) * DMODEL;

  // Staging: thread -> (k0 = tid>>6 in [0,4), n = tid&63); rows k = k0 + 4j.
  const int k0 = tid >> 6;
  const int ns = tid & 63;
  const u16* wsrc = W + (size_t)k0 * DMODEL + n0 + ns;
  u16 wreg[8];
#pragma unroll
  for (int j = 0; j < 8; ++j) wreg[j] = wsrc[(size_t)j * 4 * DMODEL];
  wsrc += (size_t)32 * DMODEL;

  for (int it = 0; it < DMODEL / 32; ++it) {
    const int p = it & 1;
    u16* btw = &bt[p][0] + ns * BTROW + k0;
#pragma unroll
    for (int j = 0; j < 8; ++j) btw[4 * j] = wreg[j];
    __syncthreads();

    if (it + 1 < DMODEL / 32) {          // overlap next slab's loads with WMMAs
#pragma unroll
      for (int j = 0; j < 8; ++j) wreg[j] = wsrc[(size_t)j * 4 * DMODEL];
      wsrc += (size_t)32 * DMODEL;
    }

    Frag16 a;
    a.q[0] = *(const uint4*)(arow + it * 32 + 16 * half);
    a.q[1] = *(const uint4*)(arow + it * 32 + 8 + 16 * half);

#pragma unroll
    for (int t = 0; t < 4; ++t) {
      const u16* br = &bt[p][0] + (t * 16 + l16) * BTROW + 16 * half;
      Frag16 bfr;
      bfr.q[0] = *(const uint4*)(br);
      bfr.q[1] = *(const uint4*)(br + 8);
      acc[t] = wmma_bf16(a, bfr, acc[t]);
    }
    // single barrier per slab: next store (parity p^1) only clobbers data
    // whose readers already passed the previous barrier.
  }

  // C layout: VGPR r -> row m0 + r + 8*half, col tile*16 + l16
#pragma unroll
  for (int t = 0; t < 4; ++t) {
#pragma unroll
    for (int r = 0; r < 8; ++r) {
      int row = m0 + r + 8 * half;
      int col = n0 + t * 16 + l16;
      float v = acc[t][r];
      if (OUTMODE == 0) {
        outB[(size_t)row * DMODEL + col] = f2bf(v);
      } else if (OUTMODE == 1) {
        int bIdx = row >> 11;            // row / S_LEN
        int tok  = row & (S_LEN - 1);
        int head = col >> 6;             // col / DKV
        int d    = col & 63;
        outB[(((size_t)(bIdx * NHEADS + head)) * DKV + d) * S_LEN + tok] = f2bf(v);
      } else {
        outF[(size_t)row * DMODEL + col] = v;
      }
    }
  }
}

// ---------------------------------------------------------------------------
// Flash attention. Block = one (b, h) x 64 queries; 4 waves, each a 16-query
// tile. K/V 32-key chunks double-buffered in LDS via CDNA5 async-to-LDS
// copies: chunk i+1 is issued before waiting on chunk i, so the DMA overlaps
// the previous chunk's WMMAs. Online softmax; row sums via an all-ones WMMA.
// ---------------------------------------------------------------------------
__global__ void __launch_bounds__(128)
attn_kernel(const u16* __restrict__ Qb,     // [b,s,h,d] bf16
            const u16* __restrict__ Kb,     // [b,s,h,d] bf16
            const u16* __restrict__ Vt,     // [b,h,d,s] bf16
            const float* __restrict__ biasT,// [H][2S-1] f32
            u16* __restrict__ ctx) {        // [b,s,h*64+d] bf16
  __shared__ u16 kbuf[2][32 * KROW];   // [buf][key][d]   5 KB each
  __shared__ u16 vbuf[2][64 * VROW];   // [buf][d][key]   5 KB each
  __shared__ u16 plds[4][16 * PROW];   // per-wave P      5 KB
  const int tid  = threadIdx.x;
  const int wave = tid >> 5;
  const int lane = tid & 31;
  const int half = lane >> 4;
  const int l16  = lane & 15;

  int blk = blockIdx.x;                   // ((b*16 + h)*32 + qb64)
  int q0  = (blk & 31) * 64 + wave * 16;
  int h   = (blk >> 5) & (NHEADS - 1);
  int b   = blk >> 9;
  const size_t vt_head = ((size_t)(b * NHEADS + h)) * DKV;

  // Q fragments for d in [0,32) and [32,64)
  Frag16 qa0, qa1;
  const u16* qrow = Qb + (size_t)(b * S_LEN + q0 + l16) * DMODEL + h * DKV;
  qa0.q[0] = *(const uint4*)(qrow + 16 * half);
  qa0.q[1] = *(const uint4*)(qrow + 8 + 16 * half);
  qa1.q[0] = *(const uint4*)(qrow + 32 + 16 * half);
  qa1.q[1] = *(const uint4*)(qrow + 40 + 16 * half);

  // all-ones B fragment (bf16 1.0) for row-sum accumulation
  Frag16 ones;
#pragma unroll
  for (int i = 0; i < 16; ++i) ones.u[i] = 0x3F80;

  f32x8 acc[4] = {f32x8{}, f32x8{}, f32x8{}, f32x8{}};  // 16q x 64d context
  f32x8 accS = {};                                       // row sums
  float run_max[8];
#pragma unroll
  for (int r = 0; r < 8; ++r) run_max[r] = -1e30f;

  u16* myp = &plds[wave][0];

  // Async copy descriptors. K: thread -> key row tid>>2, 32B chunk tid&3.
  // V: thread -> d row tid>>1, 32B chunk tid&1. Rows padded in LDS.
  unsigned klo[2] = {
    lds_off(&kbuf[0][0]) + (tid >> 2) * (KROW * 2) + (tid & 3) * 32,
    lds_off(&kbuf[1][0]) + (tid >> 2) * (KROW * 2) + (tid & 3) * 32 };
  unsigned vlo[2] = {
    lds_off(&vbuf[0][0]) + (tid >> 1) * (VROW * 2) + (tid & 1) * 32,
    lds_off(&vbuf[1][0]) + (tid >> 1) * (VROW * 2) + (tid & 1) * 32 };
  unsigned long long kg =
      (unsigned long long)(uintptr_t)(Kb + (size_t)(b * S_LEN + (tid >> 2)) * DMODEL + h * DKV)
      + (unsigned long long)((tid & 3) * 32);
  unsigned long long vg =
      (unsigned long long)(uintptr_t)(Vt + (vt_head + (tid >> 1)) * S_LEN)
      + (unsigned long long)((tid & 1) * 32);

  // Per-lane bias base: biasT[h][ (kb + key_local) - q + 2047 ]
  const float* bcur = biasT + (size_t)h * NDELTA + 2047 + l16 - q0 - 8 * half;

  // Prologue: chunk 0 -> buffer 0.
  async_b128(klo[0], kg); async_b128(klo[0] + 16, kg + 16);
  async_b128(vlo[0], vg); async_b128(vlo[0] + 16, vg + 16);
  kg += (unsigned long long)32 * DMODEL * 2;
  vg += 64;

  for (int kb = 0; kb < S_LEN; kb += 32) {
    const int p = (kb >> 5) & 1;
    if (kb + 32 < S_LEN) {
      const int nq = p ^ 1;        // issue next chunk into the other buffer
      async_b128(klo[nq], kg); async_b128(klo[nq] + 16, kg + 16);
      async_b128(vlo[nq], vg); async_b128(vlo[nq] + 16, vg + 16);
      kg += (unsigned long long)32 * DMODEL * 2;
      vg += 64;
      wait_async4();               // current chunk done; next stays in flight
    } else {
      wait_async0();
    }
    __syncthreads();

    const u16* kbp = &kbuf[p][0];
    const u16* vbp = &vbuf[p][0];

    // ---- scores: S = Q(16x64) @ K^T(64x32) as two 16x16 tiles ------------
    f32x8 s0 = {}, s1 = {};
    {
      const u16* kf0 = kbp + l16 * KROW + 16 * half;          // keys 0..15
      const u16* kf1 = kbp + (16 + l16) * KROW + 16 * half;   // keys 16..31
      Frag16 k00, k01, k10, k11;
      k00.q[0] = *(const uint4*)(kf0);       k00.q[1] = *(const uint4*)(kf0 + 8);
      k01.q[0] = *(const uint4*)(kf0 + 32);  k01.q[1] = *(const uint4*)(kf0 + 40);
      k10.q[0] = *(const uint4*)(kf1);       k10.q[1] = *(const uint4*)(kf1 + 8);
      k11.q[0] = *(const uint4*)(kf1 + 32);  k11.q[1] = *(const uint4*)(kf1 + 40);
      s0 = wmma_bf16(qa0, k00, s0);
      s0 = wmma_bf16(qa1, k01, s0);
      s1 = wmma_bf16(qa0, k10, s1);
      s1 = wmma_bf16(qa1, k11, s1);
    }

    // ---- bias + online softmax ------------------------------------------
#pragma unroll
    for (int r = 0; r < 8; ++r) {
      float v0 = s0[r] + bcur[0 - r];     // immediate-offset, coalesced
      float v1 = s1[r] + bcur[16 - r];

      float m = rowmax16(fmaxf(v0, v1));  // ds_swizzle xor-reduction

      float nm   = fmaxf(run_max[r], m);
      float corr = __expf(run_max[r] - nm);
      run_max[r] = nm;

#pragma unroll
      for (int c = 0; c < 4; ++c) acc[c][r] *= corr;
      accS[r] *= corr;

      int row = r + 8 * half;
      myp[row * PROW + l16]      = f2bf(__expf(v0 - nm));
      myp[row * PROW + l16 + 16] = f2bf(__expf(v1 - nm));
    }

    // ---- P A-fragment from wave-private LDS (C-layout -> A-layout) -------
    Frag16 pa;
    pa.q[0] = *(const uint4*)(myp + l16 * PROW + 16 * half);
    pa.q[1] = *(const uint4*)(myp + l16 * PROW + 8 + 16 * half);

    // ---- context += P @ V; row sums += P @ ones --------------------------
    const u16* vfp = vbp + l16 * VROW + 16 * half;
#pragma unroll
    for (int c = 0; c < 4; ++c) {
      Frag16 vf;
      vf.q[0] = *(const uint4*)(vfp + c * 16 * VROW);
      vf.q[1] = *(const uint4*)(vfp + c * 16 * VROW + 8);
      acc[c] = wmma_bf16(pa, vf, acc[c]);
    }
    accS = wmma_bf16(pa, ones, accS);

    __syncthreads();          // all waves done reading before buffer reuse
    bcur += 32;
  }

  // ---- normalize and store context bf16 [b,s,h*64+d] ---------------------
#pragma unroll
  for (int c = 0; c < 4; ++c) {
#pragma unroll
    for (int r = 0; r < 8; ++r) {
      int row = b * S_LEN + q0 + r + 8 * half;
      int col = h * DKV + c * 16 + l16;
      ctx[(size_t)row * DMODEL + col] = f2bf(acc[c][r] / accS[r]);
    }
  }
}

// ---------------------------------------------------------------------------
// Launch
// ---------------------------------------------------------------------------
extern "C" void kernel_launch(void* const* d_in, const int* in_sizes, int n_in,
                              void* d_out, int out_size, void* d_ws, size_t ws_size,
                              hipStream_t stream) {
  (void)in_sizes; (void)n_in; (void)out_size; (void)ws_size;
  const float* x   = (const float*)d_in[0];
  const float* wq  = (const float*)d_in[1];
  const float* wk  = (const float*)d_in[2];
  const float* wv  = (const float*)d_in[3];
  const float* wo  = (const float*)d_in[4];
  const float* rel = (const float*)d_in[5];

  // Workspace map (~48.3 MB total)
  char* ws = (char*)d_ws;
  u16*   xb    = (u16*)(ws);                         //  8 MB  x bf16 [4096,1024]
  u16*   wqb   = (u16*)(ws + ((size_t)8  << 20));    //  2 MB
  u16*   wkb   = (u16*)(ws + ((size_t)10 << 20));    //  2 MB
  u16*   wvb   = (u16*)(ws + ((size_t)12 << 20));    //  2 MB
  u16*   wob   = (u16*)(ws + ((size_t)14 << 20));    //  2 MB
  u16*   Qb    = (u16*)(ws + ((size_t)16 << 20));    //  8 MB  [b,s,h,d]
  u16*   Kb    = (u16*)(ws + ((size_t)24 << 20));    //  8 MB  [b,s,h,d]
  u16*   Vt    = (u16*)(ws + ((size_t)32 << 20));    //  8 MB  [b,h,d,s]
  u16*   ctx   = (u16*)(ws + ((size_t)40 << 20));    //  8 MB  [b,s,1024]
  float* biasT = (float*)(ws + ((size_t)48 << 20));  // 256 KB [H][2S-1]

  // 1) bf16 conversions
  cvt_f32_to_bf16<<<4096, 256, 0, stream>>>(x,  xb,  MROWS * DMODEL);
  cvt_f32_to_bf16<<<1024, 256, 0, stream>>>(wq, wqb, DMODEL * DMODEL);
  cvt_f32_to_bf16<<<1024, 256, 0, stream>>>(wk, wkb, DMODEL * DMODEL);
  cvt_f32_to_bf16<<<1024, 256, 0, stream>>>(wv, wvb, DMODEL * DMODEL);
  cvt_f32_to_bf16<<<1024, 256, 0, stream>>>(wo, wob, DMODEL * DMODEL);

  // 2) transposed relative-position bias table
  build_bias<<<dim3((NDELTA + 255) / 256, NHEADS), 256, 0, stream>>>(rel, biasT);

  // 3) QKV projections (WMMA bf16); V written transposed [b,h,d,s]
  dim3 ggrid(MROWS / 128, DMODEL / 64);
  gemm_bf16<0><<<ggrid, 256, 0, stream>>>(xb, wqb, Qb, nullptr);
  gemm_bf16<0><<<ggrid, 256, 0, stream>>>(xb, wkb, Kb, nullptr);
  gemm_bf16<1><<<ggrid, 256, 0, stream>>>(xb, wvb, Vt, nullptr);

  // 4) flash attention: 1024 blocks = (b,h,64q), 4 waves/block
  attn_kernel<<<1024, 128, 0, stream>>>(Qb, Kb, Vt, biasT, ctx);

  // 5) output projection -> f32 d_out
  gemm_bf16<2><<<ggrid, 256, 0, stream>>>(ctx, wob, nullptr, (float*)d_out);
}
